// GaussianSelfAttention_64785286693235
// MI455X (gfx1250) — matrix operand order repaired and checked by reading, not compiled, not run on hardware
//
#include <hip/hip_runtime.h>
#include <hip/hip_bf16.h>

// ---------------------------------------------------------------------------
// Gaussian relative-position self-attention for gfx1250 (MI455X, wave32).
// R (106 MB) is never read: scores are a closed-form quadratic in (k-i, l-j).
// Heavy work = 43.5 GFLOP bf16 GEMM on v_wmma_f32_16x16x32_bf16, with
// GLOBAL_LOAD_ASYNC_TO_LDS_B128 staging of the X operand.
// ---------------------------------------------------------------------------

typedef __attribute__((ext_vector_type(16))) __bf16 v16bf;
typedef __attribute__((ext_vector_type(8)))  float  v8f;
typedef unsigned short u16;
typedef unsigned int   u32;
typedef unsigned long long u64;

static constexpr int Bz = 8, Wd = 48, Hd = 48, Ed = 64, NH = 8;
static constexpr int WH = Wd * Hd;          // 2304  (K of main GEMM)
static constexpr int Mm = WH * NH;          // 18432 (M of main GEMM: (i,j,n))
static constexpr int Nn = Bz * Ed;          // 512   (N of main GEMM: (b,e))
static constexpr int M2 = Bz * WH;          // 18432 (M of out GEMM: (b,i,j))
static constexpr int K2 = NH * Ed;          // 512
static constexpr int N2 = Ed;               // 64

__device__ __forceinline__ u16 f2bf(float f) {
  u32 x = __builtin_bit_cast(u32, f);
  u32 r = x + 0x7FFFu + ((x >> 16) & 1u);   // round-to-nearest-even
  return (u16)(r >> 16);
}

union BF16Frag { v16bf v; u32 u[8]; };
union Pack16B { uint4 q; u16 s[8]; };

// LDS byte offset (wave-relative) of a __shared__ pointer, for async-LDS ops.
__device__ __forceinline__ unsigned lds_off(const void* p) {
  return (unsigned)(u64)(__attribute__((address_space(3))) const char*)p;
}

__device__ __forceinline__ float rel_score(float dx, float dy,
                                           float c0, float c1, float c2,
                                           float c3, float c4, float c5) {
  return fmaf(c0, dx, fmaf(c1, dy, fmaf(c2, dx * dx,
         fmaf(c3, dy * dy, fmaf(c4, dx * dy, c5)))));
}

// ---------------------------------------------------------------------------
// Kernel 0: pack x -> Xp_t[(b*64+e)][(kx*48+ky)] bf16 (K-contiguous rows for
// B-fragments) and W_fc -> Bpt[e_out][f] bf16.
// ---------------------------------------------------------------------------
__global__ void pack_kernel(const float* __restrict__ x,
                            const float* __restrict__ wfc,
                            u16* __restrict__ Xpt, u16* __restrict__ Bpt) {
  int idx = blockIdx.x * 256 + threadIdx.x;
  if (idx < Nn * WH) {
    int be = idx / WH, kl = idx % WH;
    int b = be >> 6, e = be & 63;
    int kx = kl / Hd, ky = kl % Hd;
    Xpt[idx] = f2bf(x[(((size_t)b * Wd + kx) * Hd + ky) * Ed + e]);
  }
  if (idx < K2 * N2) Bpt[idx] = f2bf(wfc[idx]);  // W_fc is already [e_out][f]
}

// ---------------------------------------------------------------------------
// Kernel 1: per-row (i,j,n) softmax stats (max, 1/sum) over all 2304 keys.
// ---------------------------------------------------------------------------
__global__ void stats_kernel(const float* __restrict__ Wsig,
                             const float* __restrict__ bsig,
                             float* __restrict__ rowMax,
                             float* __restrict__ rowInvZ) {
  int r = blockIdx.x * 256 + threadIdx.x;
  if (r >= Mm) return;
  int n = r & 7, ij = r >> 3, ii = ij / Hd, jj = ij % Hd;
  float c0 = Wsig[n * 5 + 0], c1 = Wsig[n * 5 + 1], c2 = Wsig[n * 5 + 2];
  float c3 = Wsig[n * 5 + 3], c4 = Wsig[n * 5 + 4], c5 = bsig[n];
  float m = -3.0e38f;
  for (int kx = 0; kx < Wd; ++kx) {
    float dx = (float)(kx - ii);
    float ax = fmaf(c2, dx * dx, fmaf(c0, dx, c5));
    float bx = fmaf(c4, dx, c1);
    for (int ky = 0; ky < Hd; ++ky) {
      float dy = (float)(ky - jj);
      m = fmaxf(m, fmaf(c3, dy * dy, fmaf(bx, dy, ax)));
    }
  }
  float sum = 0.f;
  for (int kx = 0; kx < Wd; ++kx) {
    float dx = (float)(kx - ii);
    float ax = fmaf(c2, dx * dx, fmaf(c0, dx, c5));
    float bx = fmaf(c4, dx, c1);
    for (int ky = 0; ky < Hd; ++ky) {
      float dy = (float)(ky - jj);
      sum += __expf(fmaf(c3, dy * dy, fmaf(bx, dy, ax)) - m);
    }
  }
  rowMax[r] = m;
  rowInvZ[r] = 1.0f / sum;
}

// ---------------------------------------------------------------------------
// Kernel 2: main GEMM  AH[(i,j,n)][(b,e)] = probs @ X,  M=18432 K=2304 N=512.
// Block = 64 M-rows x 256 N-cols; 8 waves, each 32Mx64N (2x4 16x16 tiles).
// probs tiles exp'ed on the fly into LDS; X chunk staged with
// GLOBAL_LOAD_ASYNC_TO_LDS_B128 (ASYNCcnt).  (kx,ky) tracked incrementally —
// no integer division in the hot loop.
// ---------------------------------------------------------------------------
__global__ __launch_bounds__(256) void attn_gemm_kernel(
    const float* __restrict__ Wsig, const float* __restrict__ bsig,
    const float* __restrict__ rowMax, const float* __restrict__ rowInvZ,
    const u16* __restrict__ Xpt, u16* __restrict__ AHp) {
  __shared__ __align__(16) u16 As[64 * 32];    // probs chunk [row][k]   4KB
  __shared__ __align__(16) u16 Bst[256 * 32];  // X chunk    [col][k]   16KB

  const int tid = threadIdx.x, lane = tid & 31, wid = tid >> 5;
  const int mrow0 = blockIdx.x * 64;
  const int col0  = blockIdx.y * 256;

  // this thread's fixed staging row for As (4 threads x 8 k per row)
  const int lr  = tid >> 2;
  const int kq0 = (tid & 3) * 8;
  const int r   = mrow0 + lr;
  const int n = r & 7, ij = r >> 3, ii = ij / Hd, jj = ij % Hd;
  const float c0 = Wsig[n * 5 + 0], c1 = Wsig[n * 5 + 1], c2 = Wsig[n * 5 + 2];
  const float c3 = Wsig[n * 5 + 3], c4 = Wsig[n * 5 + 4], c5 = bsig[n];
  const float rmv = rowMax[r], rzv = rowInvZ[r];
  const float fii = (float)ii, fjj = (float)jj;

  const int wm = wid & 1;    // M half: rows wm*32 .. wm*32+31
  const int wn = wid >> 1;   // N quarter: cols wn*64 .. wn*64+63

  // async staging addresses: this thread owns Bst row `tid` (64 contiguous B)
  const unsigned bstOff = lds_off(&Bst[tid * 32]);
  u64 gsrc = (u64)(const void*)(Xpt + (size_t)(col0 + tid) * WH);

  // incremental (kx, ky) for q0 = kq0 + 32*ks  (base of this thread's 8 keys)
  int kx0 = 0, ky0 = kq0;

  v8f acc[2][4] = {};

  for (int ks = 0; ks < WH / 32; ++ks) {
    // ---- async DMA: 64B of X per thread, global -> LDS (no VGPR round-trip)
    // inst offset applies to BOTH global and LDS addresses (ISA 10.7).
    asm volatile(
        "global_load_async_to_lds_b128 %0, %1, off\n\t"
        "global_load_async_to_lds_b128 %0, %1, off offset:16\n\t"
        "global_load_async_to_lds_b128 %0, %1, off offset:32\n\t"
        "global_load_async_to_lds_b128 %0, %1, off offset:48"
        :: "v"(bstOff), "v"(gsrc) : "memory");
    gsrc += 64;  // next 32 K-values (bf16)

    // ---- stage A chunk: probs = exp(score - max) / Z, bf16 in LDS ----
    {
      Pack16B pk;
#pragma unroll
      for (int t = 0; t < 8; ++t) {
        int ky = ky0 + t, kx = kx0;
        if (ky >= Hd) { ky -= Hd; kx += 1; }
        float dx = (float)kx - fii, dy = (float)ky - fjj;
        pk.s[t] = f2bf(__expf(rel_score(dx, dy, c0, c1, c2, c3, c4, c5) - rmv) * rzv);
      }
      *(uint4*)(As + lr * 32 + kq0) = pk.q;
      ky0 += 32;
      if (ky0 >= Hd) { ky0 -= Hd; kx0 += 1; }
    }
    asm volatile("s_wait_asynccnt 0x0" ::: "memory");
    __syncthreads();

    // ---- build fragments (contiguous k = 16*(lane/16) + s in A and B) ----
    BF16Frag a[2], b[4];
#pragma unroll
    for (int mt = 0; mt < 2; ++mt) {
      const int arow = wm * 32 + mt * 16 + (lane & 15);
      const int kb = (lane < 16) ? 0 : 16;
#pragma unroll
      for (int v = 0; v < 8; ++v) {
        int k = kb + 2 * v;
        a[mt].u[v] = *(const u32*)(As + arow * 32 + k);
      }
    }
#pragma unroll
    for (int nt = 0; nt < 4; ++nt) {
      const int nc = wn * 64 + nt * 16 + (lane & 15);
      const int kb = (lane < 16) ? 0 : 16;
#pragma unroll
      for (int v = 0; v < 8; ++v)
        b[nt].u[v] = *(const u32*)(Bst + nc * 32 + kb + 2 * v);
    }
    // ---- 8 WMMAs per wave per K-step ----
#pragma unroll
    for (int mt = 0; mt < 2; ++mt)
#pragma unroll
      for (int nt = 0; nt < 4; ++nt)
        acc[mt][nt] = __builtin_amdgcn_wmma_f32_16x16x32_bf16(
            false, a[mt].v, false, b[nt].v, (short)0, acc[mt][nt], false, false);
    __syncthreads();
  }

  // ---- store: AHp[(b*2304+ij)][(e*8+n)].  Tile M-bases are multiples of 8,
  // so the 8 accumulator rows land on consecutive n -> one 16B store each.
#pragma unroll
  for (int mt = 0; mt < 2; ++mt)
#pragma unroll
    for (int nt = 0; nt < 4; ++nt) {
      const int col = col0 + wn * 64 + nt * 16 + (lane & 15);
      const int bb = col >> 6, e = col & 63;
      const int grb = mrow0 + wm * 32 + mt * 16 + ((lane < 16) ? 0 : 8);
      const int ij2 = grb >> 3;
      Pack16B pk;
#pragma unroll
      for (int d = 0; d < 8; ++d) pk.s[d] = f2bf(acc[mt][nt][d]);
      *(uint4*)(AHp + ((size_t)bb * WH + ij2) * 512 + e * 8) = pk.q;
    }
}

// ---------------------------------------------------------------------------
// Kernel 3: out[b,i,j,:] = AH[b,i,j,:] @ W_fc^T + b_fc.  M=18432 K=512 N=64.
// Pure global->WMMA (128-bit fragment loads), fused bias, f32 output.
// ---------------------------------------------------------------------------
__global__ __launch_bounds__(256) void out_gemm_kernel(
    const u16* __restrict__ AHp, const u16* __restrict__ Bpt,
    const float* __restrict__ bfc, float* __restrict__ out) {
  const int tid = threadIdx.x, lane = tid & 31, wid = tid >> 5;
  const int row = blockIdx.x * 128 + wid * 16 + (lane & 15);
  const int hi = (lane < 16) ? 0 : 1;
  v8f acc[4] = {};

  for (int ks = 0; ks < K2 / 32; ++ks) {
    const int kb = ks * 32;
    BF16Frag a;
    const u16* ap = AHp + (size_t)row * K2 + kb + hi * 8;
    *(uint4*)&a.u[0] = *(const uint4*)ap;          // K = kb+base .. +7
    *(uint4*)&a.u[4] = *(const uint4*)(ap + 16);   // K = kb+16+base .. +7
#pragma unroll
    for (int nt = 0; nt < 4; ++nt) {
      BF16Frag b;
      const u16* bp = Bpt + (size_t)(nt * 16 + (lane & 15)) * K2 + kb + hi * 8;
      *(uint4*)&b.u[0] = *(const uint4*)bp;        // K = kb+base .. +7
      *(uint4*)&b.u[4] = *(const uint4*)(bp + 16); // K = kb+16+base .. +7
      acc[nt] = __builtin_amdgcn_wmma_f32_16x16x32_bf16(
          false, a.v, false, b.v, (short)0, acc[nt], false, false);
    }
  }
#pragma unroll
  for (int nt = 0; nt < 4; ++nt) {
    int col = nt * 16 + (lane & 15);
    float bias = bfc[col];
#pragma unroll
    for (int d = 0; d < 8; ++d) {
      int gr = blockIdx.x * 128 + wid * 16 + hi * 8 + d;
      out[(size_t)gr * N2 + col] = acc[nt][d] + bias;
    }
  }
}

// ---------------------------------------------------------------------------
// Launch.  Inputs: 0=hidden_states 1=attention_mask(unused) 2=R(UNUSED!)
//                  3=W_sigma 4=b_sigma 5=W_fc 6=b_fc
// Workspace: rowMax f32[18432] | rowInvZ f32[18432] | Bpt bf16[512*64]
//            | Xp_t bf16[512*2304] | AHp bf16[18432*512]   (~20.5 MB total)
// ---------------------------------------------------------------------------
extern "C" void kernel_launch(void* const* d_in, const int* in_sizes, int n_in,
                              void* d_out, int out_size, void* d_ws, size_t ws_size,
                              hipStream_t stream) {
  const float* x    = (const float*)d_in[0];
  const float* Wsig = (const float*)d_in[3];
  const float* bsig = (const float*)d_in[4];
  const float* wfc  = (const float*)d_in[5];
  const float* bfc  = (const float*)d_in[6];
  float* out = (float*)d_out;

  float* rowMax  = (float*)d_ws;
  float* rowInvZ = rowMax + Mm;
  u16*   Bpt     = (u16*)(rowInvZ + Mm);
  u16*   Xpt     = Bpt + (size_t)K2 * N2;
  u16*   AHp     = Xpt + (size_t)Nn * WH;

  pack_kernel<<<(Nn * WH + 255) / 256, 256, 0, stream>>>(x, wfc, Xpt, Bpt);
  stats_kernel<<<Mm / 256, 256, 0, stream>>>(Wsig, bsig, rowMax, rowInvZ);
  attn_gemm_kernel<<<dim3(Mm / 64, Nn / 256), 256, 0, stream>>>(
      Wsig, bsig, rowMax, rowInvZ, Xpt, AHp);
  out_gemm_kernel<<<M2 / 128, 256, 0, stream>>>(AHp, Bpt, bfc, out);
}
